// DGRACL_45930380263725
// MI455X (gfx1250) — compile-verified, matrix operand order checked by reading, not run on hardware
//
#include <hip/hip_runtime.h>
#include <hip/hip_bf16.h>

// ---------------------------------------------------------------------------
// DGRACL retrieval-augmented anomaly score, fused for MI455X (gfx1250).
//   BQ=2048 queries, BN=65536 pool, H=256, K=7.
// v3: __launch_bounds__(256,1) relaxes the VGPR cap so the A fragments
// (async-staged into LDS, then hoisted to registers by the compiler) no
// longer spill to scratch inside the hot loop; per-tile address math is
// strength-reduced to pointer increments.
// ---------------------------------------------------------------------------

#define BQ 2048
#define BN 65536
#define H  256
#define TOPK 7
#define NCHUNK 32                 // pool split across grid.y
#define CHUNK (BN / NCHUNK)       // 2048 columns / block
#define WCOLS (CHUNK / 8)         // 256 columns / wave
#define NTILE (WCOLS / 16)        // 16 N-tiles of 16 cols per wave
#define ASTRIDE 264               // padded LDS row stride (halves): 528B
#define LAMBDA_LOG2E (-0.14426950408889634f)   // -0.1 * log2(e)
#define BIGSHIFT 1.0e4f
#define SCALE 0.0625f             // H^-0.5 = 1/16

typedef _Float16 v16h __attribute__((ext_vector_type(16)));
typedef _Float16 v8h  __attribute__((ext_vector_type(8)));
typedef float    v8f  __attribute__((ext_vector_type(8)));

union HFrag { v16h v; v8h h[2]; };

__device__ __forceinline__ float wave_sum(float x) {
#pragma unroll
  for (int off = 16; off >= 1; off >>= 1) x += __shfl_xor(x, off, 32);
  return x;
}

// Async copy of one 16-byte chunk global -> LDS (gfx1250 async-to-LDS engine,
// tracked by ASYNCcnt). lds_dst low 32 bits are the LDS byte offset.
__device__ __forceinline__ void async_copy16_to_lds(void* lds_dst, const void* gsrc) {
  unsigned ldsa = (unsigned)(uintptr_t)lds_dst;
  asm volatile("global_load_async_to_lds_b128 %0, %1, off"
               :: "v"(ldsa), "v"(gsrc) : "memory");
}

__device__ __forceinline__ void wait_asynccnt0() {
  asm volatile("s_wait_asynccnt 0" ::: "memory");
}

// Sorted (descending) register-resident top-7 insertion network.
__device__ __forceinline__ void topk7_insert(float (&v)[TOPK], int (&id)[TOPK],
                                             float val, int idx) {
#pragma unroll
  for (int i = TOPK - 1; i >= 1; --i) {
    bool shift = val > v[i - 1];   // candidate ranks above slot i-1 -> old i-1 moves down
    bool place = val > v[i];       // candidate lands at slot i
    float nv = shift ? v[i - 1] : (place ? val : v[i]);
    int   ni = shift ? id[i - 1] : (place ? idx : id[i]);
    v[i] = nv; id[i] = ni;
  }
  bool p0 = val > v[0];
  v[0]  = p0 ? val : v[0];
  id[0] = p0 ? idx : id[0];
}

// ---------------------------------------------------------------------------
// Kernel 0: L2-normalize rows of pool_emb / query_emb into f16. 1 wave / row.
// ---------------------------------------------------------------------------
__global__ void dgracl_normalize_kernel(const float* __restrict__ pool_emb,
                                        const float* __restrict__ query_emb,
                                        _Float16* __restrict__ pn,
                                        _Float16* __restrict__ qn) {
  int row  = blockIdx.x * 8 + (threadIdx.x >> 5);
  int lane = threadIdx.x & 31;
  if (row >= BN + BQ) return;
  const float* src; _Float16* dst;
  if (row < BN) { src = pool_emb  + (size_t)row * H;        dst = pn + (size_t)row * H; }
  else          { src = query_emb + (size_t)(row - BN) * H; dst = qn + (size_t)(row - BN) * H; }
  float4 a = *(const float4*)(src + lane * 8);
  float4 b = *(const float4*)(src + lane * 8 + 4);
  float ss = a.x*a.x + a.y*a.y + a.z*a.z + a.w*a.w
           + b.x*b.x + b.y*b.y + b.z*b.z + b.w*b.w;
  ss = wave_sum(ss);
  float rn = 1.0f / fmaxf(sqrtf(ss), 1e-12f);
  v8h o;
  o[0]=(_Float16)(a.x*rn); o[1]=(_Float16)(a.y*rn);
  o[2]=(_Float16)(a.z*rn); o[3]=(_Float16)(a.w*rn);
  o[4]=(_Float16)(b.x*rn); o[5]=(_Float16)(b.y*rn);
  o[6]=(_Float16)(b.z*rn); o[7]=(_Float16)(b.w*rn);
  *(v8h*)(dst + lane * 8) = o;
}

// ---------------------------------------------------------------------------
// Kernel 1: fused WMMA similarity + decay + causal mask + top-7 partials.
// grid = (BQ/16, NCHUNK), block = 256 (8 waves). Wave w owns 256 pool cols.
// __launch_bounds__(256,1): allow a full VGPR budget so the 8 A fragments
// (64 VGPRs) stay resident -- no scratch spills in the K loop.
// ---------------------------------------------------------------------------
__global__ void __launch_bounds__(256, 1)
dgracl_sim_topk_kernel(const float* __restrict__ query_time,
                       const float* __restrict__ pool_time,
                       const _Float16* __restrict__ qn,
                       const _Float16* __restrict__ pn,
                       float* __restrict__ part_val,
                       int*   __restrict__ part_idx) {
  __shared__ __align__(16) _Float16 atile[16 * ASTRIDE]; // shared A tile (padded)
  __shared__ __align__(16) float ctile[8 * 256];         // per-wave 16x16 C tile
  __shared__ float pv[16 * 16 * TOPK];                   // [row][list][k]
  __shared__ int   pi[16 * 16 * TOPK];

  const int qtile = blockIdx.x;
  const int chunk = blockIdx.y;
  const int wave  = threadIdx.x >> 5;
  const int lane  = threadIdx.x & 31;
  const int l15   = lane & 15;
  const int hi    = lane >> 4;
  const int qbase = qtile * 16;

  // --- stage A tile (16 rows x 256 halves) into LDS with async copies.
  // 512 x 16B chunks, 2 per thread; each wave waits its own ASYNCcnt, then
  // the workgroup barrier makes the full tile visible to everyone.
  {
    int c0 = threadIdx.x;          // chunks c0 and c0+256
    int r0 = c0 >> 5, o0 = c0 & 31;
    async_copy16_to_lds(atile + r0 * ASTRIDE + o0 * 8,
                        qn + (size_t)(qbase + r0) * H + o0 * 8);
    int c1 = c0 + 256;
    int r1 = c1 >> 5, o1 = c1 & 31;
    async_copy16_to_lds(atile + r1 * ASTRIDE + o1 * 8,
                        qn + (size_t)(qbase + r1) * H + o1 * 8);
  }
  wait_asynccnt0();
  __syncthreads();

  // Selection role: lane owns query row (lane>>1), columns 8*(lane&1)..+7.
  const int rowSel = lane >> 1;
  const int hsel   = lane & 1;
  const float qtv  = query_time[qbase + rowSel];

  float tv[TOPK]; int ti[TOPK];
#pragma unroll
  for (int k = 0; k < TOPK; ++k) { tv[k] = -__builtin_inff(); ti[k] = 0; }

  const int colwave = chunk * CHUNK + wave * WCOLS;
  // 16-bit A layout: lane m=l15, hi selects K sub-halves 8*hi..+7 and +16.
  const _Float16* aptr = atile + l15 * ASTRIDE + 8 * hi;

  // Steady-state pointers, incremented per tile (no per-tile mul chains).
  const _Float16* brow = pn + (size_t)(colwave + l15) * H + 8 * hi;
  const float*    tptr = pool_time + colwave + 8 * hsel;
  int             cidx = colwave + 8 * hsel;

  for (int t = 0; t < NTILE; ++t) {
    // Prefetch next tile's B rows into cache (global_prefetch_b8).
    if (t + 1 < NTILE)
      __builtin_prefetch(brow + (size_t)16 * H, 0, 1);

    // --- 16x16 sim tile, K=256 via 8 WMMAs (f32 accumulate).
    // A fragments from LDS (hoisted to registers), B fragments from global.
    v8f c = {};
#pragma unroll
    for (int kk = 0; kk < 8; ++kk) {
      HFrag a, b;
      a.h[0] = *(const v8h*)(aptr + kk * 32);
      a.h[1] = *(const v8h*)(aptr + kk * 32 + 16);
      b.h[0] = *(const v8h*)(brow + kk * 32);
      b.h[1] = *(const v8h*)(brow + kk * 32 + 16);
      c = __builtin_amdgcn_wmma_f32_16x16x32_f16(
          false, a.v, false, b.v, (short)0, c, false, false);
    }

    // --- stage C to LDS (row-major 16x16) for row-owned selection.
    float* cw = ctile + wave * 256;
#pragma unroll
    for (int r = 0; r < 8; ++r)
      cw[(r + 8 * hi) * 16 + l15] = c[r];

    // --- decay + causal shift + top-7 (same wave wrote LDS; hw waits dscnt).
    const float4* cp = (const float4*)(cw + rowSel * 16 + 8 * hsel);
    const float4* tp = (const float4*)tptr;
    float4 s0 = cp[0], s1 = cp[1];
    float4 p0 = tp[0], p1 = tp[1];
    float sv[8] = { s0.x, s0.y, s0.z, s0.w, s1.x, s1.y, s1.z, s1.w };
    float pt[8] = { p0.x, p0.y, p0.z, p0.w, p1.x, p1.y, p1.z, p1.w };
#pragma unroll
    for (int j = 0; j < 8; ++j) {
      float d   = fabsf(qtv - pt[j]);
      float w   = exp2f(d * LAMBDA_LOG2E);              // exp(-0.1*|dt|)
      // Non-causal entries shifted by -1e4: causal always wins if any exists,
      // else relative order of unmasked sim_t is preserved (fallback row).
      float val = sv[j] * w + ((pt[j] < qtv) ? 0.0f : -BIGSHIFT);
      topk7_insert(tv, ti, val, cidx + j);
    }

    brow += (size_t)16 * H;
    tptr += 16;
    cidx += 16;
  }

  // --- per-block merge: 16 partial lists per row -> one list per row.
  {
    int listIdx = wave * 2 + hsel;
    int base = (rowSel * 16 + listIdx) * TOPK;
#pragma unroll
    for (int k = 0; k < TOPK; ++k) { pv[base + k] = tv[k]; pi[base + k] = ti[k]; }
  }
  __syncthreads();

  if (threadIdx.x < 16) {
    int row = threadIdx.x;
    float mv[TOPK]; int mi[TOPK];
#pragma unroll
    for (int k = 0; k < TOPK; ++k) { mv[k] = -__builtin_inff(); mi[k] = 0; }
    for (int l = 0; l < 16; ++l) {
      int base = (row * 16 + l) * TOPK;
      for (int k = 0; k < TOPK; ++k)
        topk7_insert(mv, mi, pv[base + k], pi[base + k]);
    }
    size_t out = ((size_t)(qtile * NCHUNK + chunk) * 16 + row) * TOPK;
    for (int k = 0; k < TOPK; ++k) { part_val[out + k] = mv[k]; part_idx[out + k] = mi[k]; }
  }
}

// ---------------------------------------------------------------------------
// Kernel 2: merge NCHUNK partial lists per query -> final top-7 indices.
// ---------------------------------------------------------------------------
__global__ void dgracl_merge_kernel(const float* __restrict__ part_val,
                                    const int*   __restrict__ part_idx,
                                    int* __restrict__ top_idx) {
  int q = blockIdx.x * 256 + threadIdx.x;
  if (q >= BQ) return;
  int qtile = q >> 4, row = q & 15;
  float mv[TOPK]; int mi[TOPK];
#pragma unroll
  for (int k = 0; k < TOPK; ++k) { mv[k] = -__builtin_inff(); mi[k] = 0; }
  for (int c = 0; c < NCHUNK; ++c) {
    size_t base = ((size_t)(qtile * NCHUNK + c) * 16 + row) * TOPK;
    for (int k = 0; k < TOPK; ++k)
      topk7_insert(mv, mi, part_val[base + k], part_idx[base + k]);
  }
  for (int k = 0; k < TOPK; ++k) top_idx[q * TOPK + k] = mi[k];
}

// ---------------------------------------------------------------------------
// Kernel 3: cross-attention fusion + anomaly score. One wave per query.
// ---------------------------------------------------------------------------
__global__ void dgracl_fuse_score_kernel(const float* __restrict__ query_emb,
                                         const float* __restrict__ pool_emb,
                                         const int*   __restrict__ top_idx,
                                         float* __restrict__ out) {
  int q    = blockIdx.x * 8 + (threadIdx.x >> 5);
  int lane = threadIdx.x & 31;
  if (q >= BQ) return;

  const float* qrow = query_emb + (size_t)q * H + lane * 8;
  float4 q0 = *(const float4*)qrow;
  float4 q1 = *(const float4*)(qrow + 4);
  float qv[8] = { q0.x, q0.y, q0.z, q0.w, q1.x, q1.y, q1.z, q1.w };

  float r[TOPK][8];
  float logit[TOPK];
#pragma unroll
  for (int k = 0; k < TOPK; ++k) {
    int idx = top_idx[q * TOPK + k];
    const float* prow = pool_emb + (size_t)idx * H + lane * 8;
    float4 a = *(const float4*)prow;
    float4 b = *(const float4*)(prow + 4);
    r[k][0]=a.x; r[k][1]=a.y; r[k][2]=a.z; r[k][3]=a.w;
    r[k][4]=b.x; r[k][5]=b.y; r[k][6]=b.z; r[k][7]=b.w;
    float d = 0.f;
#pragma unroll
    for (int j = 0; j < 8; ++j) d += qv[j] * r[k][j];
    logit[k] = wave_sum(d) * SCALE;
  }

  float m = logit[0];
#pragma unroll
  for (int k = 1; k < TOPK; ++k) m = fmaxf(m, logit[k]);
  float den = 0.f, e[TOPK];
#pragma unroll
  for (int k = 0; k < TOPK; ++k) { e[k] = expf(logit[k] - m); den += e[k]; }
  float inv = 1.0f / den;

  float f[8] = {0,0,0,0,0,0,0,0};
#pragma unroll
  for (int k = 0; k < TOPK; ++k) {
    float a = e[k] * inv;
#pragma unroll
    for (int j = 0; j < 8; ++j) f[j] += a * r[k][j];
  }

  float dqf = 0.f, qq = 0.f, ff = 0.f, dd = 0.f;
#pragma unroll
  for (int j = 0; j < 8; ++j) {
    dqf += qv[j] * f[j];
    qq  += qv[j] * qv[j];
    ff  += f[j] * f[j];
    float t = qv[j] - f[j];
    dd  += t * t;
  }
  dqf = wave_sum(dqf); qq = wave_sum(qq); ff = wave_sum(ff); dd = wave_sum(dd);

  float cosv  = dqf / fmaxf(sqrtf(qq) * sqrtf(ff), 1e-8f);
  float score = 0.5f * (1.0f - cosv) + 0.5f * sqrtf(dd);
  if (lane == 0) out[q] = score;
}

// ---------------------------------------------------------------------------
extern "C" void kernel_launch(void* const* d_in, const int* in_sizes, int n_in,
                              void* d_out, int out_size, void* d_ws, size_t ws_size,
                              hipStream_t stream) {
  const float* query_emb  = (const float*)d_in[0];   // [BQ*H]
  const float* query_time = (const float*)d_in[1];   // [BQ]
  const float* pool_emb   = (const float*)d_in[2];   // [BN*H]
  const float* pool_time  = (const float*)d_in[3];   // [BN]
  float* out = (float*)d_out;                        // [BQ]

  // Workspace layout (256B aligned sections, ~38.3 MB total).
  char* w = (char*)d_ws;
  _Float16* pn       = (_Float16*)(w);                      // 33,554,432 B
  _Float16* qn       = (_Float16*)(w + 33554432);           //  1,048,576 B
  float*    part_val = (float*)   (w + 34603008);           //  1,835,008 B
  int*      part_idx = (int*)     (w + 36438016);           //  1,835,008 B
  int*      top_idx  = (int*)     (w + 38273024);           //     57,344 B

  // 0: normalize -> f16
  dgracl_normalize_kernel<<<(BN + BQ) / 8, 256, 0, stream>>>(
      pool_emb, query_emb, pn, qn);

  // 1: fused WMMA similarity + decay/mask + partial top-7
  dgracl_sim_topk_kernel<<<dim3(BQ / 16, NCHUNK), 256, 0, stream>>>(
      query_time, pool_time, qn, pn, part_val, part_idx);

  // 2: merge partials -> final top-7 indices
  dgracl_merge_kernel<<<BQ / 256, 256, 0, stream>>>(part_val, part_idx, top_idx);

  // 3: fusion + anomaly score
  dgracl_fuse_score_kernel<<<BQ / 8, 256, 0, stream>>>(
      query_emb, pool_emb, top_idx, out);
}
